// E2ESPDMBlock_18717467476432
// MI455X (gfx1250) — matrix operand order repaired and checked by reading, not compiled, not run on hardware
//
#include <hip/hip_runtime.h>
#include <hip/hip_bf16.h>
#include <math.h>

// ---------------------------------------------------------------------------
// CDNA5 (gfx1250) wave32 WMMA types
// ---------------------------------------------------------------------------
typedef __bf16 v16bf __attribute__((ext_vector_type(16)));
typedef float  v8f   __attribute__((ext_vector_type(8)));

#define WMMA_BF16(a, b, c) \
  __builtin_amdgcn_wmma_f32_16x16x32_bf16(false, (a), false, (b), (short)0, (c), false, false)

// Problem constants (from reference setup_inputs)
constexpr int BB = 4, T = 1024, D = 1024, F = 4096, S = 4096, NF = 16, H = 16, HD = 64;
constexpr int MROWS = BB * T;  // 4096

__device__ __forceinline__ float gelu_f(float x) {
  return 0.5f * x * (1.0f + erff(x * 0.70710678118654752f));
}

// ---------------------------------------------------------------------------
// LayerNorm: one 256-thread block per row of D=1024
// ---------------------------------------------------------------------------
__global__ void ln_kernel(const float* __restrict__ x, const float* __restrict__ g,
                          const float* __restrict__ b, float* __restrict__ out) {
  const int row = blockIdx.x;
  const int tid = threadIdx.x;
  const float* xr = x + (size_t)row * D;
  __shared__ float sh[256];
  float s = 0.f, ss = 0.f;
  for (int i = tid; i < D; i += 256) { float v = xr[i]; s += v; ss += v * v; }
  sh[tid] = s; __syncthreads();
  for (int st = 128; st; st >>= 1) { if (tid < st) sh[tid] += sh[tid + st]; __syncthreads(); }
  const float mu = sh[0] / (float)D;
  __syncthreads();
  sh[tid] = ss; __syncthreads();
  for (int st = 128; st; st >>= 1) { if (tid < st) sh[tid] += sh[tid + st]; __syncthreads(); }
  const float var = sh[0] / (float)D - mu * mu;
  const float inv = rsqrtf(var + 1e-5f);
  __syncthreads();
  for (int i = tid; i < D; i += 256)
    out[(size_t)row * D + i] = (xr[i] - mu) * inv * g[i] + b[i];
}

// ---------------------------------------------------------------------------
// Generic WMMA GEMM: C[M,N] = act(A[M,K] @ W[N,K]^T + bias (+ resid))
// Block = 256 threads (8 waves), macro-tile 128x128, K-step 32.
// Each wave computes a 32x64 sub-tile (2x4 WMMA fragments of 16x16).
// fp32 -> bf16 conversion on the fly; float4 global loads into LDS.
// ---------------------------------------------------------------------------
template <int ACT>
__global__ __launch_bounds__(256)
void gemm_wmma_kernel(const float* __restrict__ A, const float* __restrict__ W,
                      const float* __restrict__ bias, const float* __restrict__ resid,
                      float* __restrict__ C, int M, int N, int K) {
  constexpr int BM = 128, BN = 128, BK = 32, LDT = 36;
  __shared__ __bf16 As[BM * LDT];
  __shared__ __bf16 Bs[BN * LDT];
  const int tid  = threadIdx.x;
  const int lane = tid & 31;
  const int wave = tid >> 5;
  const int laneHalf = lane >> 4;
  const int lmod = lane & 15;
  const int wrow = wave & 3;   // 4 row bands of 32
  const int wcol = wave >> 2;  // 2 col bands of 64
  const int m0 = blockIdx.y * BM, n0 = blockIdx.x * BN;

  v8f acc[2][4] = {};

  // loader mapping: 256 threads cover 128 rows x 32 cols in 4 passes of float4
  const int lr = tid >> 3;        // 0..31
  const int lc = (tid & 7) * 4;   // 0..28 step 4

  for (int k0 = 0; k0 < K; k0 += BK) {
    __syncthreads();
#pragma unroll
    for (int p = 0; p < 4; ++p) {
      const int r = lr + p * 32;
      const float4 av = *(const float4*)&A[(size_t)(m0 + r) * K + k0 + lc];
      const float4 wv = *(const float4*)&W[(size_t)(n0 + r) * K + k0 + lc];
      __bf16* ad = &As[r * LDT + lc];
      __bf16* bd = &Bs[r * LDT + lc];
      ad[0] = (__bf16)av.x; ad[1] = (__bf16)av.y; ad[2] = (__bf16)av.z; ad[3] = (__bf16)av.w;
      bd[0] = (__bf16)wv.x; bd[1] = (__bf16)wv.y; bd[2] = (__bf16)wv.z; bd[3] = (__bf16)wv.w;
    }
    __syncthreads();

    v16bf af[2], bfr[4];
#pragma unroll
    for (int mt = 0; mt < 2; ++mt) {
      const __bf16* Ar = &As[(wrow * 32 + mt * 16 + lmod) * LDT];
#pragma unroll
      for (int i = 0; i < 8; ++i) {
        af[mt][i]     = Ar[laneHalf * 8 + i];
        af[mt][8 + i] = Ar[16 + laneHalf * 8 + i];
      }
    }
#pragma unroll
    for (int nt = 0; nt < 4; ++nt) {
      const __bf16* Br = &Bs[(wcol * 64 + nt * 16 + lmod) * LDT];
#pragma unroll
      for (int i = 0; i < 8; ++i) { bfr[nt][i] = Br[laneHalf * 8 + i]; bfr[nt][8 + i] = Br[16 + laneHalf * 8 + i]; }
    }
#pragma unroll
    for (int mt = 0; mt < 2; ++mt)
#pragma unroll
      for (int nt = 0; nt < 4; ++nt)
        acc[mt][nt] = WMMA_BF16(af[mt], bfr[nt], acc[mt][nt]);
  }

#pragma unroll
  for (int mt = 0; mt < 2; ++mt)
#pragma unroll
    for (int nt = 0; nt < 4; ++nt)
#pragma unroll
      for (int r = 0; r < 8; ++r) {
        const int m = m0 + wrow * 32 + mt * 16 + r + laneHalf * 8;
        const int n = n0 + wcol * 64 + nt * 16 + lmod;
        float v = acc[mt][nt][r];
        if (bias)  v += bias[n];
        if (resid) v += resid[(size_t)m * N + n];
        if (ACT == 1) v = gelu_f(v);
        C[(size_t)m * N + n] = v;
      }
}

// ---------------------------------------------------------------------------
// Flash-style causal attention, 1 wave per (b, h, 16 q-rows). head_dim = 64.
// qkv layout: [B,T,3D], q at +h*64, k at +D+h*64, v at +2D+h*64.
// ---------------------------------------------------------------------------
__global__ void attn_kernel(const float* __restrict__ qkv, float* __restrict__ out) {
  __shared__ __bf16 Ps[16 * 34];
  const int lane = threadIdx.x & 31;
  const int laneHalf = lane >> 4;
  const int lmod = lane & 15;
  const int qb = blockIdx.x * 16;
  const int b = blockIdx.y / H, h = blockIdx.y % H;
  const size_t base = ((size_t)b * T) * 3 * D + (size_t)h * HD;

  // Q A-fragments (16x64 split into two K=32 chunks)
  v16bf aq[2];
  {
    const float* qr = qkv + base + (size_t)(qb + lmod) * 3 * D;
#pragma unroll
    for (int c = 0; c < 2; ++c)
#pragma unroll
      for (int i = 0; i < 8; ++i) {
        aq[c][i]     = (__bf16)qr[c * 32 + laneHalf * 8 + i];
        aq[c][8 + i] = (__bf16)qr[c * 32 + 16 + laneHalf * 8 + i];
      }
  }

  v8f o[4] = {};
  float mrow[8], lrow[8];
#pragma unroll
  for (int r = 0; r < 8; ++r) { mrow[r] = -3.0e38f; lrow[r] = 0.f; }

  for (int kb = 0; kb < qb + 16; kb += 32) {
    // S = Q @ K^T for 32 keys -> two 16x16 tiles
    v8f s[2];
#pragma unroll
    for (int nt = 0; nt < 2; ++nt) {
      v16bf bk0, bk1;
      const float* kr = qkv + base + D + (size_t)(kb + nt * 16 + lmod) * 3 * D;
#pragma unroll
      for (int i = 0; i < 8; ++i) {
        bk0[i]     = (__bf16)kr[laneHalf * 8 + i];
        bk0[8 + i] = (__bf16)kr[16 + laneHalf * 8 + i];
        bk1[i]     = (__bf16)kr[32 + laneHalf * 8 + i];
        bk1[8 + i] = (__bf16)kr[48 + laneHalf * 8 + i];
      }
      v8f z = {};
      z = WMMA_BF16(aq[0], bk0, z);
      z = WMMA_BF16(aq[1], bk1, z);
      s[nt] = z;
    }

    // online softmax per row (row = r + 8*laneHalf in C layout)
#pragma unroll
    for (int r = 0; r < 8; ++r) {
      const int qg = qb + r + laneHalf * 8;
      float v0 = s[0][r] * 0.125f; if (kb + lmod > qg)      v0 = -3.0e38f;
      float v1 = s[1][r] * 0.125f; if (kb + 16 + lmod > qg) v1 = -3.0e38f;
      float mx = fmaxf(v0, v1);
#pragma unroll
      for (int off = 1; off < 16; off <<= 1) mx = fmaxf(mx, __shfl_xor(mx, off, 32));
      const float mn = fmaxf(mrow[r], mx);
      const float sc = __expf(mrow[r] - mn);
      const float p0 = __expf(v0 - mn), p1 = __expf(v1 - mn);
      float ls = p0 + p1;
#pragma unroll
      for (int off = 1; off < 16; off <<= 1) ls += __shfl_xor(ls, off, 32);
      lrow[r] = lrow[r] * sc + ls;
      mrow[r] = mn;
#pragma unroll
      for (int dt = 0; dt < 4; ++dt) o[dt][r] *= sc;
      Ps[(r + laneHalf * 8) * 34 + lmod]      = (__bf16)p0;
      Ps[(r + laneHalf * 8) * 34 + 16 + lmod] = (__bf16)p1;
    }
    __syncthreads();

    // P (16x32) as A-fragment, V (32x64) as B-fragments
    v16bf ap;
    {
      const __bf16* pr = &Ps[lmod * 34];
#pragma unroll
      for (int i = 0; i < 8; ++i) {
        ap[i]     = pr[laneHalf * 8 + i];
        ap[8 + i] = pr[16 + laneHalf * 8 + i];
      }
    }
    const float* Vb = qkv + base + 2 * D;
#pragma unroll
    for (int dt = 0; dt < 4; ++dt) {
      v16bf bv;
#pragma unroll
      for (int i = 0; i < 8; ++i) {
        bv[i]     = (__bf16)Vb[(size_t)(kb + laneHalf * 8 + i) * 3 * D + dt * 16 + lmod];
        bv[8 + i] = (__bf16)Vb[(size_t)(kb + 16 + laneHalf * 8 + i) * 3 * D + dt * 16 + lmod];
      }
      o[dt] = WMMA_BF16(ap, bv, o[dt]);
    }
    __syncthreads();
  }

#pragma unroll
  for (int dt = 0; dt < 4; ++dt)
#pragma unroll
    for (int r = 0; r < 8; ++r) {
      const int qg = qb + r + laneHalf * 8;
      out[((size_t)b * T + qg) * D + h * HD + dt * 16 + lmod] = o[dt][r] / lrow[r];
    }
}

// ---------------------------------------------------------------------------
// proto_basis = family_proj_w (16,S) @ W_dict (S,D) -> (16,D). Tiny GEMM.
// ---------------------------------------------------------------------------
__global__ void proto_kernel(const float* __restrict__ fp, const float* __restrict__ Wd,
                             float* __restrict__ proto) {
  const int o = blockIdx.x * 256 + threadIdx.x;  // 0..16383
  const int f = o >> 10, d = o & 1023;
  float s = 0.f;
  for (int k = 0; k < S; ++k) s += fp[f * S + k] * Wd[(size_t)k * D + d];
  proto[o] = s;
}

// ---------------------------------------------------------------------------
// Per row: family softmax -> prototypes -> residual = h - prototypes
// ---------------------------------------------------------------------------
__global__ void dictres_kernel(const float* __restrict__ hx, const float* __restrict__ keys,
                               const float* __restrict__ proto, float* __restrict__ resid) {
  const int row = blockIdx.x, tid = threadIdx.x;
  const float* xr = hx + (size_t)row * D;
  __shared__ float red[256];
  __shared__ float sc[NF];
  __shared__ float probs[NF];
  float xv[4];
#pragma unroll
  for (int j = 0; j < 4; ++j) xv[j] = xr[tid + j * 256];
  for (int f = 0; f < NF; ++f) {
    float p = 0.f;
#pragma unroll
    for (int j = 0; j < 4; ++j) p += xv[j] * keys[f * D + tid + j * 256];
    red[tid] = p; __syncthreads();
    for (int st = 128; st; st >>= 1) { if (tid < st) red[tid] += red[tid + st]; __syncthreads(); }
    if (tid == 0) sc[f] = red[0];
    __syncthreads();
  }
  if (tid == 0) {
    float mx = sc[0];
    for (int f = 1; f < NF; ++f) mx = fmaxf(mx, sc[f]);
    float sum = 0.f;
    for (int f = 0; f < NF; ++f) { probs[f] = __expf(sc[f] - mx); sum += probs[f]; }
    for (int f = 0; f < NF; ++f) probs[f] /= sum;
  }
  __syncthreads();
#pragma unroll
  for (int j = 0; j < 4; ++j) {
    const int d = tid + j * 256;
    float pv = 0.f;
    for (int f = 0; f < NF; ++f) pv += probs[f] * proto[f * D + d];
    resid[(size_t)row * D + d] = xv[j] - pv;
  }
}

// ---------------------------------------------------------------------------
// Top-8 of 4096 per row, offset = sparse @ W_dict, accumulate losses
// ---------------------------------------------------------------------------
__global__ void topk_kernel(const float* __restrict__ coeffs, const float* __restrict__ resid,
                            const float* __restrict__ Wd, float* __restrict__ losses) {
  const int row = blockIdx.x, tid = threadIdx.x;
  __shared__ float cv[S];
  __shared__ float rv[256];
  __shared__ int   ri[256];
  __shared__ float tv[8];
  __shared__ int   tix[8];
  for (int j = tid; j < S; j += 256) cv[j] = coeffs[(size_t)row * S + j];
  __syncthreads();
  for (int r = 0; r < 8; ++r) {
    float bm = -3.4e38f; int bi = 0;
    for (int j = tid; j < S; j += 256) if (cv[j] > bm) { bm = cv[j]; bi = j; }
    rv[tid] = bm; ri[tid] = bi; __syncthreads();
    for (int st = 128; st; st >>= 1) {
      if (tid < st && rv[tid + st] > rv[tid]) { rv[tid] = rv[tid + st]; ri[tid] = ri[tid + st]; }
      __syncthreads();
    }
    if (tid == 0) { tv[r] = rv[0]; tix[r] = ri[0]; cv[ri[0]] = -3.4e38f; }
    __syncthreads();
  }
  float lsum = 0.f;
#pragma unroll
  for (int j = 0; j < 4; ++j) {
    const int d = tid + j * 256;
    float off = 0.f;
#pragma unroll
    for (int r = 0; r < 8; ++r) off += tv[r] * Wd[(size_t)tix[r] * D + d];
    const float df = off - resid[(size_t)row * D + d];
    lsum += df * df;
  }
  rv[tid] = lsum; __syncthreads();
  for (int st = 128; st; st >>= 1) { if (tid < st) rv[tid] += rv[tid + st]; __syncthreads(); }
  if (tid == 0) {
    atomicAdd(&losses[0], rv[0] / ((float)MROWS * (float)D));
    float sp = 0.f;
    for (int r = 0; r < 8; ++r) sp += fabsf(tv[r]);
    atomicAdd(&losses[1], sp / ((float)MROWS * (float)S));
  }
}

// ---------------------------------------------------------------------------
// LIF scan over T, one thread per (b, f); coalesced along F
// ---------------------------------------------------------------------------
__global__ void lif_kernel(const float* __restrict__ gate, const float* __restrict__ theta,
                           unsigned char* __restrict__ spikes) {
  const int idx = blockIdx.x * 256 + threadIdx.x;  // 0..B*F-1
  const int b = idx / F, f = idx % F;
  float mem = 0.f;
  const float th = theta[f];
  const float* g = gate + (size_t)b * T * F + f;
  unsigned char* sp = spikes + (size_t)b * T * F + f;
  for (int t = 0; t < T; ++t) {
    mem = 0.8f * mem + g[(size_t)t * F];
    const unsigned char s = (mem > 1.0f) ? 1 : 0;
    mem -= th * (float)s;
    sp[(size_t)t * F] = s;
  }
}

// ---------------------------------------------------------------------------
// Per-row cosine between l2norm(mlp_act) and l2norm(spikes); accumulate 1-mean
// ---------------------------------------------------------------------------
__global__ void cos_kernel(const float* __restrict__ act, const unsigned char* __restrict__ spikes,
                           float* __restrict__ losses) {
  const int row = blockIdx.x, tid = threadIdx.x;
  const float* ar = act + (size_t)row * F;
  const unsigned char* sr = spikes + (size_t)row * F;
  float a2 = 0.f, dot = 0.f, cnt = 0.f;
  for (int j = tid; j < F; j += 256) {
    const float a = ar[j];
    const float s = (float)sr[j];
    a2 += a * a; dot += a * s; cnt += s;
  }
  __shared__ float red[256];
  __shared__ float res3[3];
  float vals[3] = {a2, dot, cnt};
  for (int k = 0; k < 3; ++k) {
    red[tid] = vals[k]; __syncthreads();
    for (int st = 128; st; st >>= 1) { if (tid < st) red[tid] += red[tid + st]; __syncthreads(); }
    if (tid == 0) res3[k] = red[0];
    __syncthreads();
  }
  if (tid == 0) {
    const float na = fmaxf(sqrtf(res3[0]), 1e-12f);
    const float ns = fmaxf(sqrtf(res3[2]), 1e-12f);
    const float c = res3[1] / (na * ns);
    atomicAdd(&losses[2], (1.0f - c) / (float)MROWS);
  }
}

__global__ void zero_losses_kernel(float* __restrict__ losses) {
  if (threadIdx.x < 3) losses[threadIdx.x] = 0.f;
}

// ---------------------------------------------------------------------------
// Launch
// ---------------------------------------------------------------------------
extern "C" void kernel_launch(void* const* d_in, const int* in_sizes, int n_in,
                              void* d_out, int out_size, void* d_ws, size_t ws_size,
                              hipStream_t stream) {
  (void)in_sizes; (void)n_in; (void)out_size; (void)ws_size;
  const float* x     = (const float*)d_in[0];
  const float* ln1_g = (const float*)d_in[1];
  const float* ln1_b = (const float*)d_in[2];
  const float* inw   = (const float*)d_in[3];
  const float* inb   = (const float*)d_in[4];
  const float* outw  = (const float*)d_in[5];
  const float* outb  = (const float*)d_in[6];
  const float* ln2_g = (const float*)d_in[7];
  const float* ln2_b = (const float*)d_in[8];
  const float* mw1   = (const float*)d_in[9];
  const float* mb1   = (const float*)d_in[10];
  const float* mw2   = (const float*)d_in[11];
  const float* mb2   = (const float*)d_in[12];
  const float* fkeys = (const float*)d_in[13];
  const float* fproj = (const float*)d_in[14];
  const float* Wd    = (const float*)d_in[15];
  const float* ow1   = (const float*)d_in[16];
  const float* ob1   = (const float*)d_in[17];
  const float* ow2   = (const float*)d_in[18];
  const float* ob2   = (const float*)d_in[19];
  const float* lifw  = (const float*)d_in[20];
  const float* theta = (const float*)d_in[21];

  float* out_x  = (float*)d_out;
  float* losses = out_x + (size_t)MROWS * D;  // [recon, sparsity, lif]

  // Workspace arena (floats), with reuse
  float* ws      = (float*)d_ws;
  float* normed  = ws;                       // 4,194,304  (LN1 then LN2)
  float* qkvb    = ws + 4194304;             // 12,582,912 (qkv)
  float* h1      = qkvb;                     // reuse (dead after attention)
  unsigned char* spikes = (unsigned char*)qkvb;  // reuse (dead after off_w2 gemm)
  float* attn_t  = ws + 16777216;            // 4,194,304  (attn out -> then residual)
  float* residb  = attn_t;                   // reuse
  float* x1      = ws + 20971520;            // 4,194,304
  float* coeffs  = ws + 25165824;            // 16,777,216
  float* gate    = coeffs;                   // reuse (dead after topk)
  float* mlp_act = ws + 41943040;            // 16,777,216
  float* proto   = ws + 58720256;            // 16,384

  zero_losses_kernel<<<1, 32, 0, stream>>>(losses);

  // 1) LN1, qkv projection
  ln_kernel<<<MROWS, 256, 0, stream>>>(x, ln1_g, ln1_b, normed);
  gemm_wmma_kernel<0><<<dim3((3 * D) / 128, MROWS / 128), 256, 0, stream>>>(
      normed, inw, inb, nullptr, qkvb, MROWS, 3 * D, D);

  // 2) causal flash attention
  attn_kernel<<<dim3(T / 16, BB * H), 32, 0, stream>>>(qkvb, attn_t);

  // 3) out-proj + residual -> x1
  gemm_wmma_kernel<0><<<dim3(D / 128, MROWS / 128), 256, 0, stream>>>(
      attn_t, outw, outb, x, x1, MROWS, D, D);

  // 4) dict constraint
  proto_kernel<<<64, 256, 0, stream>>>(fproj, Wd, proto);
  dictres_kernel<<<MROWS, 256, 0, stream>>>(x1, fkeys, proto, residb);
  gemm_wmma_kernel<1><<<dim3(D / 128, MROWS / 128), 256, 0, stream>>>(
      residb, ow1, ob1, nullptr, h1, MROWS, D, D);
  gemm_wmma_kernel<0><<<dim3(S / 128, MROWS / 128), 256, 0, stream>>>(
      h1, ow2, ob2, nullptr, coeffs, MROWS, S, D);
  topk_kernel<<<MROWS, 256, 0, stream>>>(coeffs, residb, Wd, losses);

  // 5) LN2, MLP, final residual -> d_out
  ln_kernel<<<MROWS, 256, 0, stream>>>(x1, ln2_g, ln2_b, normed);
  gemm_wmma_kernel<1><<<dim3(F / 128, MROWS / 128), 256, 0, stream>>>(
      normed, mw1, mb1, nullptr, mlp_act, MROWS, F, D);
  gemm_wmma_kernel<0><<<dim3(D / 128, MROWS / 128), 256, 0, stream>>>(
      mlp_act, mw2, mb2, x1, out_x, MROWS, D, F);

  // 6) LIF constraint
  gemm_wmma_kernel<0><<<dim3(F / 128, MROWS / 128), 256, 0, stream>>>(
      normed, lifw, nullptr, nullptr, gate, MROWS, F, D);
  lif_kernel<<<(BB * F) / 256, 256, 0, stream>>>(gate, theta, spikes);
  cos_kernel<<<MROWS, 256, 0, stream>>>(mlp_act, spikes, losses);
}
